// NerfRender_29197187678876
// MI455X (gfx1250) — compile-verified, matrix operand order checked by reading, not run on hardware
//
#include <hip/hip_runtime.h>
#include <hip/hip_bf16.h>
#include <math.h>

typedef _Float16 h16;
typedef __attribute__((ext_vector_type(16))) _Float16 v16h;
typedef __attribute__((ext_vector_type(8)))  _Float16 v8h;
typedef __attribute__((ext_vector_type(8)))  float    v8f;

#define NEAR_P 2.0f
#define FAR_P  6.0f
#define N_COARSE 64
#define S_FINE   192   // 64 uniform + 128 importance samples

struct MlpPtrs {
  const h16*   xyz_w[8];
  const float* xyz_b[8];
  const h16* feat_w; const float* feat_b;
  const h16* dir_w;  const float* dir_b;
  const h16* sig_w;  const float* sig_b;
  const h16* rgb_w;  const float* rgb_b;
};

// ---------------------------------------------------------------------------
// Weight packing: f32 -> f16 in WMMA B-fragment order.
// Fragment (ntile, kstep) = 32x16 f16 tile; lane l holds 16 contiguous halves:
//   col = ntile*16 + (l&15); K(i) = kstep*32 + (l>>4)*8 + (i&7) + (i>>3)*16
// perm==1: dir-layer row permutation so input = [feat(256) | dir_emb(27) | 0pad]
// ---------------------------------------------------------------------------
__global__ void pack_weights_kernel(const float* __restrict__ src, h16* __restrict__ dst,
                                    int K, int N, int Kpad, int Npad, int perm, int total)
{
  const int ksteps = Kpad >> 5;
  for (int idx = blockIdx.x * blockDim.x + threadIdx.x; idx < total;
       idx += gridDim.x * blockDim.x) {
    const int i    = idx & 15;
    const int lane = (idx >> 4) & 31;
    const int frag = idx >> 9;
    const int nt = frag / ksteps;
    const int ks = frag - nt * ksteps;
    const int col = (nt << 4) + (lane & 15);
    const int k   = (ks << 5) + ((lane >> 4) << 3) + (i & 7) + ((i >> 3) << 4);
    int sk = perm ? ((k < 256) ? (k + 27) : (k - 256)) : k;
    const bool okk = perm ? (k < 283) : (sk < K);
    float v = 0.f;
    if (okk && col < N) v = src[(size_t)sk * N + col];
    dst[idx] = (h16)v;
  }
}

__global__ void cvt_f16_kernel(const float* __restrict__ src, h16* __restrict__ dst, int n)
{
  int i = blockIdx.x * blockDim.x + threadIdx.x;
  if (i < n) dst[i] = (h16)src[i];
}

// ---------------------------------------------------------------------------
// 64-row x nOut GEMM on LDS activations, in-place safe:
// all waves finish reading (K loop) -> barrier -> bias/ReLU writeback -> barrier
// wave w owns columns [w*32, w*32+32); 4 M-tiles x 2 N-tiles of f32 accum.
// ---------------------------------------------------------------------------
__device__ __forceinline__ void gemm64(const h16* __restrict__ in, int inStride, int ksteps,
                                       h16* __restrict__ out,
                                       const h16* __restrict__ wp,
                                       const float* __restrict__ bias,
                                       int nOut, bool relu)
{
  const int lane  = threadIdx.x & 31;
  const int wv    = threadIdx.x >> 5;
  const int n0    = wv * 32;
  const int lrow  = lane & 15;
  const int lhalf = lane >> 4;
  const bool act  = (n0 < nOut);

  v8f acc[4][2];
  const v8f vzero = {0.f, 0.f, 0.f, 0.f, 0.f, 0.f, 0.f, 0.f};
#pragma unroll
  for (int m = 0; m < 4; ++m)
#pragma unroll
    for (int n = 0; n < 2; ++n) acc[m][n] = vzero;

  if (act) {
    for (int ks = 0; ks < ksteps; ++ks) {
      const int kb = ks << 5;
      v16h bfrag[2];
#pragma unroll
      for (int n = 0; n < 2; ++n) {
        const int ntg = (n0 >> 4) + n;
        const h16* bp = wp + (((size_t)(ntg * ksteps + ks)) << 9) + (lane << 4);
        bfrag[n] = *(const v16h*)bp;           // 32B contiguous: packed B fragment
      }
#pragma unroll
      for (int m = 0; m < 4; ++m) {
        const int row = (m << 4) + lrow;
        const h16* ap = in + (size_t)row * inStride + kb + (lhalf << 3);
        union { v16h v; v8h h[2]; } af;
        af.h[0] = *(const v8h*)ap;             // K = kb + o .. +7
        af.h[1] = *(const v8h*)(ap + 16);      // K = kb + 16 + o .. +7
#pragma unroll
        for (int n = 0; n < 2; ++n)
          acc[m][n] = __builtin_amdgcn_wmma_f32_16x16x32_f16(
              false, af.v, false, bfrag[n], (short)0, acc[m][n], false, false);
      }
    }
  }
  __syncthreads();   // all reads of `in` complete before any in-place write
  if (act) {
#pragma unroll
    for (int n = 0; n < 2; ++n) {
      const int col = n0 + (n << 4) + lrow;
      const float bv = bias[col];
#pragma unroll
      for (int m = 0; m < 4; ++m) {
#pragma unroll
        for (int v = 0; v < 8; ++v) {
          float x = acc[m][n][v] + bv;
          if (relu) x = fmaxf(x, 0.f);
          out[(size_t)((m << 4) + (lhalf << 3) + v) * 320 + col] = (h16)x;
        }
      }
    }
  }
  __syncthreads();
}

// ---------------------------------------------------------------------------
// Fully fused NeRF MLP: 64 samples per block, 8 wave32.
// ---------------------------------------------------------------------------
__global__ __launch_bounds__(256)
void nerf_mlp_kernel(const float* __restrict__ rays_o, const float* __restrict__ rays_d,
                     const float* __restrict__ zbuf, int n_rays, int S, int total,
                     MlpPtrs P, float* __restrict__ sigmaOut, float* __restrict__ rgbOut)
{
  __shared__ h16 sAct[64 * 320];   // 40 KB activation ping buffer (K-padded)
  __shared__ h16 sEmb[64 * 64];    //  8 KB xyz positional encoding (63 + pad)
  __shared__ h16 sDirE[64 * 28];   //  ~4 KB direction encoding (27 + pad)

  const int tid = threadIdx.x;
  const int gs  = blockIdx.x * 64;

  // --- stage 0: positional + direction encodings (threads 0..63, 1 row each)
  if (tid < 64) {
    int s = gs + tid; if (s >= total) s = total - 1;
    const int ray = s / S, si = s - ray * S;
    const float ox = rays_o[3 * ray], oy = rays_o[3 * ray + 1], oz = rays_o[3 * ray + 2];
    const float dx = rays_d[3 * ray], dy = rays_d[3 * ray + 1], dz = rays_d[3 * ray + 2];
    float z = zbuf ? zbuf[(size_t)ray * S + si]
                   : NEAR_P + (FAR_P - NEAR_P) * ((float)si / (float)(S - 1));
    float p[3] = {ox + dx * z, oy + dy * z, oz + dz * z};
    h16* e = sEmb + tid * 64;
    e[0] = (h16)p[0]; e[1] = (h16)p[1]; e[2] = (h16)p[2];
    float f = 1.f;
    for (int fr = 0; fr < 10; ++fr) {
#pragma unroll
      for (int c = 0; c < 3; ++c) {
        e[3 + 6 * fr + c]     = (h16)__sinf(f * p[c]);
        e[3 + 6 * fr + 3 + c] = (h16)__cosf(f * p[c]);
      }
      f *= 2.f;
    }
    e[63] = (h16)0.f;
    const float inorm = 1.f / (sqrtf(dx * dx + dy * dy + dz * dz) + 1e-5f);
    float v[3] = {dx * inorm, dy * inorm, dz * inorm};
    h16* de = sDirE + tid * 28;
    de[0] = (h16)v[0]; de[1] = (h16)v[1]; de[2] = (h16)v[2];
    f = 1.f;
    for (int fr = 0; fr < 4; ++fr) {
#pragma unroll
      for (int c = 0; c < 3; ++c) {
        de[3 + 6 * fr + c]     = (h16)__sinf(f * v[c]);
        de[3 + 6 * fr + 3 + c] = (h16)__cosf(f * v[c]);
      }
      f *= 2.f;
    }
    de[27] = (h16)0.f;
  }
  __syncthreads();

  // --- layer 0: emb(64) -> 256
  gemm64(sEmb, 64, 2, sAct, P.xyz_w[0], P.xyz_b[0], 256, true);

  // --- layers 1..7 (skip-concat at layer 4: cols 256..319 <- emb)
  for (int L = 1; L < 8; ++L) {
    if (L == 4) {
      for (int idx = tid; idx < 64 * 64; idx += 256) {
        const int rr = idx >> 6, cc = idx & 63;
        sAct[rr * 320 + 256 + cc] = sEmb[rr * 64 + cc];
      }
      __syncthreads();
    }
    gemm64(sAct, 320, (L == 4) ? 10 : 8, sAct, P.xyz_w[L], P.xyz_b[L], 256, true);
  }

  // --- sigma head: relu(h . w + b), one thread per row
  if (tid < 64) {
    int s = gs + tid; if (s >= total) s = total - 1;
    float a = P.sig_b[0];
    const h16* hrow = sAct + tid * 320;
#pragma unroll 8
    for (int k = 0; k < 256; ++k) a += (float)hrow[k] * (float)P.sig_w[k];
    sigmaOut[s] = fmaxf(a, 0.f);
  }
  __syncthreads();

  // --- feature layer: 256 -> 256, no relu, in place
  gemm64(sAct, 320, 8, sAct, P.feat_w, P.feat_b, 256, false);

  // --- assemble dir input: [feat(0..255) | dir_emb(256..282) | 0(283..319)]
  for (int idx = tid; idx < 64 * 64; idx += 256) {
    const int rr = idx >> 6, cc = idx & 63;
    sAct[rr * 320 + 256 + cc] = (cc < 27) ? sDirE[rr * 28 + cc] : (h16)0.f;
  }
  __syncthreads();

  // --- dir layer: 288 -> 128, relu, in place (waves 0..3 active)
  gemm64(sAct, 320, 9, sAct, P.dir_w, P.dir_b, 128, true);

  // --- rgb head: sigmoid(h2 . w + b)
  if (tid < 64) {
    int s = gs + tid; if (s >= total) s = total - 1;
    const h16* hrow = sAct + tid * 320;
#pragma unroll
    for (int c = 0; c < 3; ++c) {
      float a = P.rgb_b[c];
#pragma unroll 8
      for (int k = 0; k < 128; ++k) a += (float)hrow[k] * (float)P.rgb_w[k * 3 + c];
      rgbOut[(size_t)s * 3 + c] = 1.f / (1.f + __expf(-a));
    }
  }
}

// ---------------------------------------------------------------------------
// Volumetric rendering: thread per ray, exclusive-cumprod transmittance.
// ---------------------------------------------------------------------------
__global__ void volrender_kernel(const float* __restrict__ rays_d, const float* __restrict__ zbuf,
                                 const float* __restrict__ sigma, const float* __restrict__ rgb,
                                 int n_rays, int S, float* __restrict__ out, int outBase,
                                 float* __restrict__ wOut)
{
  const int r = blockIdx.x * blockDim.x + threadIdx.x;
  if (r >= n_rays) return;
  const float dx = rays_d[3 * r], dy = rays_d[3 * r + 1], dz = rays_d[3 * r + 2];
  const float dn = sqrtf(dx * dx + dy * dy + dz * dz);
  float T = 1.f, cr = 0.f, cg = 0.f, cb = 0.f, depth = 0.f;
  for (int i = 0; i < S; ++i) {
    const float zi = zbuf ? zbuf[(size_t)r * S + i]
                          : NEAR_P + (FAR_P - NEAR_P) * ((float)i / (float)(S - 1));
    float delta;
    if (i < S - 1) {
      const float zn = zbuf ? zbuf[(size_t)r * S + i + 1]
                            : NEAR_P + (FAR_P - NEAR_P) * ((float)(i + 1) / (float)(S - 1));
      delta = (zn - zi) * dn;
    } else delta = 1e10f * dn;
    const float sg = sigma[(size_t)r * S + i];
    float a = 1.f - __expf(-delta * fmaxf(sg, 0.f));
    if (i == S - 1) a = 1.f;
    const float w = a * T;
    T *= (1.f - a + 1e-10f);
    const size_t b = ((size_t)r * S + i) * 3;
    cr += w * rgb[b]; cg += w * rgb[b + 1]; cb += w * rgb[b + 2];
    depth += w * zi;
    if (wOut) wOut[(size_t)r * S + i] = w;
  }
  out[(size_t)r * 8 + outBase + 0] = cr;
  out[(size_t)r * 8 + outBase + 1] = cg;
  out[(size_t)r * 8 + outBase + 2] = cb;
  out[(size_t)r * 8 + outBase + 3] = depth;
}

// ---------------------------------------------------------------------------
// Deterministic inverse-CDF sampling + O(n) sorted merge (both lists sorted).
// ---------------------------------------------------------------------------
__global__ void pdf_merge_kernel(const float* __restrict__ wC, int n_rays,
                                 float* __restrict__ zall)
{
  const int r = blockIdx.x * blockDim.x + threadIdx.x;
  if (r >= n_rays) return;
  float cdf[63];
  float wloc[62];
  float wsum = 0.f;
  for (int j = 0; j < 62; ++j) {
    const float w = wC[(size_t)r * 64 + 1 + j] + 1e-5f;
    wloc[j] = w; wsum += w;
  }
  cdf[0] = 0.f;
  float c = 0.f;
  for (int j = 0; j < 62; ++j) { c += wloc[j] / wsum; cdf[j + 1] = c; }
  const float step = (FAR_P - NEAR_P) / 63.f;   // z(i)=NEAR+step*i ; mid(i)=NEAR+step*(i+0.5)

  auto sample = [&](int j) -> float {
    const float u = (float)j / 127.f;
    int lo = 0, hi = 63;                        // searchsorted right over 63-entry cdf
    while (lo < hi) { const int mid = (lo + hi) >> 1; if (cdf[mid] <= u) lo = mid + 1; else hi = mid; }
    int below = lo - 1; if (below < 0) below = 0;
    int above = lo;     if (above > 62) above = 62;
    const float cg0 = cdf[below], cg1 = cdf[above];
    const float bg0 = NEAR_P + step * ((float)below + 0.5f);
    const float bg1 = NEAR_P + step * ((float)above + 0.5f);
    float den = cg1 - cg0; if (den < 1e-5f) den = 1.f;
    return bg0 + ((u - cg0) / den) * (bg1 - bg0);
  };

  int ui = 0, sj = 0;
  float zs = sample(0);
  float* zo = zall + (size_t)r * 192;
  for (int k = 0; k < 192; ++k) {
    const float zu = NEAR_P + (FAR_P - NEAR_P) * ((float)ui / 63.f);
    const bool takeU = (ui < 64) && (sj >= 128 || zu <= zs);
    if (takeU) { zo[k] = zu; ++ui; }
    else       { zo[k] = zs; ++sj; if (sj < 128) zs = sample(sj); }
  }
}

// ---------------------------------------------------------------------------
extern "C" void kernel_launch(void* const* d_in, const int* in_sizes, int n_in,
                              void* d_out, int out_size, void* d_ws, size_t ws_size,
                              hipStream_t stream)
{
  const float* rays_o = (const float*)d_in[0];
  const float* rays_d = (const float*)d_in[1];
  const int n_rays = in_sizes[0] / 3;
  const int NC = n_rays * N_COARSE;
  const int NF = n_rays * S_FINE;

  char* ws = (char*)d_ws;
  size_t off = 0;
  auto alloc = [&](size_t bytes) -> char* {
    char* p = ws + off; off += (bytes + 255) & ~(size_t)255; return p;
  };

  float* sigma_c = (float*)alloc((size_t)NC * 4);
  float* rgb_c   = (float*)alloc((size_t)NC * 12);
  float* w_c     = (float*)alloc((size_t)NC * 4);
  float* z_all   = (float*)alloc((size_t)NF * 4);
  float* sigma_f = (float*)alloc((size_t)NF * 4);
  float* rgb_f   = (float*)alloc((size_t)NF * 12);

  auto packW = [&](const float* src, int K, int N, int Kpad, int Npad, int perm) -> const h16* {
    const int total = (Kpad / 32) * (Npad / 16) * 512;
    h16* dst = (h16*)alloc((size_t)total * 2);
    pack_weights_kernel<<<(total + 255) / 256, 256, 0, stream>>>(src, dst, K, N, Kpad, Npad, perm, total);
    return dst;
  };
  auto packV = [&](const float* src, int n) -> const h16* {
    h16* dst = (h16*)alloc((size_t)n * 2);
    cvt_f16_kernel<<<(n + 255) / 256, 256, 0, stream>>>(src, dst, n);
    return dst;
  };

  // Param flattening (jax pytree, sorted keys): per MLP:
  // dir(w,b), feature(w,b), rgb(w,b), sigma(w,b), xyz0..7(w,b)  -> 24 arrays
  MlpPtrs P[2];
  for (int m = 0; m < 2; ++m) {
    const int b = 2 + m * 24;
    const float* dir_w  = (const float*)d_in[b + 0];
    const float* dir_b  = (const float*)d_in[b + 1];
    const float* feat_w = (const float*)d_in[b + 2];
    const float* feat_b = (const float*)d_in[b + 3];
    const float* rgb_w  = (const float*)d_in[b + 4];
    const float* rgb_b  = (const float*)d_in[b + 5];
    const float* sig_w  = (const float*)d_in[b + 6];
    const float* sig_b  = (const float*)d_in[b + 7];
    for (int i = 0; i < 8; ++i) {
      const float* xw = (const float*)d_in[b + 8 + 2 * i];
      const float* xb = (const float*)d_in[b + 9 + 2 * i];
      const int K  = (i == 0) ? 63 : (i == 4) ? 319 : 256;
      const int Kp = (i == 0) ? 64 : (i == 4) ? 320 : 256;
      P[m].xyz_w[i] = packW(xw, K, 256, Kp, 256, 0);
      P[m].xyz_b[i] = xb;
    }
    P[m].feat_w = packW(feat_w, 256, 256, 256, 256, 0); P[m].feat_b = feat_b;
    P[m].dir_w  = packW(dir_w, 283, 128, 288, 128, 1);  P[m].dir_b  = dir_b;
    P[m].sig_w  = packV(sig_w, 256);                    P[m].sig_b  = sig_b;
    P[m].rgb_w  = packV(rgb_w, 128 * 3);                P[m].rgb_b  = rgb_b;
  }
  (void)n_in; (void)out_size; (void)ws_size;

  float* out = (float*)d_out;

  // coarse MLP (uniform z) -> sigma_c, rgb_c
  nerf_mlp_kernel<<<NC / 64, 256, 0, stream>>>(rays_o, rays_d, nullptr, n_rays, N_COARSE, NC,
                                               P[0], sigma_c, rgb_c);
  // coarse render (+ per-sample weights for PDF)
  volrender_kernel<<<(n_rays + 255) / 256, 256, 0, stream>>>(rays_d, nullptr, sigma_c, rgb_c,
                                                             n_rays, N_COARSE, out, 0, w_c);
  // importance sampling + sorted merge -> z_all[2048 x 192]
  pdf_merge_kernel<<<(n_rays + 255) / 256, 256, 0, stream>>>(w_c, n_rays, z_all);
  // fine MLP
  nerf_mlp_kernel<<<NF / 64, 256, 0, stream>>>(rays_o, rays_d, z_all, n_rays, S_FINE, NF,
                                               P[1], sigma_f, rgb_f);
  // fine render
  volrender_kernel<<<(n_rays + 255) / 256, 256, 0, stream>>>(rays_d, z_all, sigma_f, rgb_f,
                                                             n_rays, S_FINE, out, 4, nullptr);
}
